// AiAModule_45818711113845
// MI455X (gfx1250) — compile-verified, matrix operand order checked by reading, not run on hardware
//
#include <hip/hip_runtime.h>
#include <hip/hip_bf16.h>

typedef __attribute__((ext_vector_type(16))) __bf16 v16bf;
typedef __attribute__((ext_vector_type(8)))  float  v8f;

#define DEV __device__ __forceinline__

static constexpr int E = 256, H = 8, HD = 32, B = 32, L = 400, MD = 64, FS = 400;
static constexpr int BH = B * H;     // 256
static constexpr int LL = L * L;     // 160000
static constexpr int WPB = 8;        // wave32 waves per 256-thread block

// ---------------- wave helpers ----------------
DEV int wave_id() { return blockIdx.x * WPB + (threadIdx.x >> 5); }

DEV float wsum(float x) {
#pragma unroll
  for (int m = 16; m; m >>= 1) x += __shfl_xor(x, m, 32);
  return x;
}
DEV float wmaxr(float x) {
#pragma unroll
  for (int m = 16; m; m >>= 1) x = fmaxf(x, __shfl_xor(x, m, 32));
  return x;
}

// ---------------- WMMA fragment helpers (wave32, 16x16x32 bf16) ----------------
// A-fragment (16-bit, 16x32): lanes 0-15 = rows 0-15 / K 0-7,16-23; lanes 16-31
// same rows, K+8.  B loaded as B^T rows (every GEMM here is X@W^T or Q@K^T).
DEV v16bf load_frag(const float* __restrict__ base, int rs, int ks, int k0, int K) {
  const int l = threadIdx.x & 31;
  const int r = l & 15, hi = l >> 4;
  v16bf f;
#pragma unroll
  for (int e = 0; e < 16; ++e) {
    int k = k0 + ((e < 8) ? e : (e + 8)) + hi * 8;
    float x = (k < K) ? base[r * rs + k * ks] : 0.0f;
    f[e] = (__bf16)x;
  }
  return f;
}

DEV v8f wmma_acc(v16bf a, v16bf b, v8f c) {
  return __builtin_amdgcn_wmma_f32_16x16x32_bf16(false, a, false, b, (short)0, c,
                                                 false, false);
}

// 16x(NT*16) tile GEMM: one A fragment reused against NT B fragments per K-step.
template <int NT>
DEV void gemm_nt(const float* __restrict__ A, int ars, int aks,
                 const float* __restrict__ Bt, int brs, int bks, int K,
                 v8f (&acc)[NT]) {
  for (int k0 = 0; k0 < K; k0 += 32) {
    v16bf a = load_frag(A, ars, aks, k0, K);
#pragma unroll
    for (int j = 0; j < NT; ++j) {
      v16bf b = load_frag(Bt + j * 16 * brs, brs, bks, k0, K);
      acc[j] = wmma_acc(a, b, acc[j]);
    }
  }
}

// C/D layout: VGPR i -> row (hi*8+i), col = lane&15.
DEV void store_tile(const v8f& c, float* base, int ms) {
  const int l = threadIdx.x & 31;
  const int n = l & 15, mb = (l >> 4) * 8;
#pragma unroll
  for (int i = 0; i < 8; ++i) base[(mb + i) * ms + n] = c[i];
}
DEV void store_tile_bs(const v8f& c, float* base, int ms, const float* bias, int n0,
                       float scale) {
  const int l = threadIdx.x & 31;
  const int n = l & 15, mb = (l >> 4) * 8;
  const float bv = bias[n0 + n];
#pragma unroll
  for (int i = 0; i < 8; ++i) base[(mb + i) * ms + n] = (c[i] + bv) * scale;
}

// ---------------- K1: fused in-projection  dst[bh,l,d] = (src[l,b,:]@W^T+b)*scale
// waves: B * 25(m) * 4(ngroup of 64 cols) = 3200
__global__ void k_inproj(const float* __restrict__ src, const float* __restrict__ W,
                         const float* __restrict__ bias, float* __restrict__ dst,
                         float scale) {
  int w = wave_id();
  if (w >= B * 25 * 4) return;
  int b = w / 100, rem = w % 100, mt = rem / 4, ng = rem % 4;
  const float* A  = src + (mt * 16) * (B * E) + b * E;   // rows over l
  const float* Bt = W + (ng * 64) * E;
  v8f acc[4] = {};
  gemm_nt<4>(A, B * E, 1, Bt, E, 1, E, acc);
#pragma unroll
  for (int j = 0; j < 4; ++j) {
    int e0 = ng * 64 + j * 16;
    int h = e0 >> 5, dlo = e0 & 31;
    float* out = dst + ((b * H + h) * L + mt * 16) * HD + dlo;
    store_tile_bs(acc[j], out, HD, bias, e0, scale);
  }
}

// ---------------- K2: attn[bh] = q[bh] @ k[bh]^T   (400x400x32)
// waves: BH * 25(m) * 7(ngroup) = 44800; last n-group has 1 valid tile.
__global__ void k_qkt(const float* __restrict__ q, const float* __restrict__ k,
                      float* __restrict__ attn) {
  int w = wave_id();
  if (w >= BH * 25 * 7) return;
  int bh = w / 175, rem = w % 175, mt = rem / 7, ng = rem % 7;
  const float* A = q + bh * (L * HD) + mt * 16 * HD;
  v8f acc[4] = {};
  for (int k0 = 0; k0 < HD; k0 += 32) {
    v16bf a = load_frag(A, HD, 1, k0, HD);
#pragma unroll
    for (int j = 0; j < 4; ++j) {
      if (ng * 4 + j < 25) {
        v16bf b = load_frag(k + bh * (L * HD) + (ng * 64 + j * 16) * HD, HD, 1, k0, HD);
        acc[j] = wmma_acc(a, b, acc[j]);
      }
    }
  }
#pragma unroll
  for (int j = 0; j < 4; ++j)
    if (ng * 4 + j < 25)
      store_tile(acc[j], attn + bh * LL + mt * 16 * L + (ng * 64 + j * 16), L);
}

// ---------------- K3a: qk_raw[(s*BH+bh), md] = cm[s,bh,:] @ Wc^T + cb
// cm[s,bh,t] = attn[bh,t,s]; N=64 exact. waves: BH*25 = 6400
__global__ void k_corrproj(const float* __restrict__ attn, const float* __restrict__ Wc,
                           const float* __restrict__ cb, float* __restrict__ qk) {
  int w = wave_id();
  if (w >= BH * 25) return;
  int bh = w / 25, mt = w % 25;
  const float* A = attn + bh * LL + mt * 16;            // row=s ->1, k=t ->L
  v8f acc[4] = {};
  gemm_nt<4>(A, 1, L, Wc, FS, 1, FS, acc);
#pragma unroll
  for (int j = 0; j < 4; ++j)
    store_tile_bs(acc[j], qk + ((mt * 16) * BH + bh) * MD + j * 16, BH * MD, cb,
                  j * 16, 1.0f);
}

// ---------------- K3c: ln2 stats over t of attn[bh,t,s] per (bh,s)
__global__ void k_ln2stats(const float* __restrict__ attn, float* __restrict__ mean2,
                           float* __restrict__ rstd2) {
  int w = wave_id();
  if (w >= BH * L) return;
  int bh = w / L, s = w % L;
  const float* col = attn + bh * LL + s;
  int l = threadIdx.x & 31;
  float sum = 0.f, sq = 0.f;
  for (int t = l; t < L; t += 32) { float x = col[t * L]; sum += x; sq += x * x; }
  sum = wsum(sum); sq = wsum(sq);
  float m = sum * (1.0f / L), v = sq * (1.0f / L) - m * m;
  if (l == 0) { mean2[w] = m; rstd2[w] = rsqrtf(v + 1e-5f); }
}

// ---------------- K3b: qk = LN1(qk_raw) + pos   (row of 64, wave per row)
__global__ void k_ln1pos(float* __restrict__ qk, const float* __restrict__ g1,
                         const float* __restrict__ b1, const float* __restrict__ pos) {
  int w = wave_id();
  if (w >= L * BH) return;                // row = s*BH + bh
  int s = w >> 8, bh = w & 255, b = bh >> 3;
  float* row = qk + (long)w * MD;
  int l = threadIdx.x & 31;
  float x0 = row[l], x1 = row[l + 32];
  float sum = wsum(x0 + x1), sq = wsum(x0 * x0 + x1 * x1);
  float m = sum * (1.0f / MD), v = sq * (1.0f / MD) - m * m;
  float rs = rsqrtf(v + 1e-5f);
  const float* p = pos + (s * B + b) * MD;
  row[l]      = (x0 - m) * rs * g1[l]      + b1[l]      + p[l];
  row[l + 32] = (x1 - m) * rs * g1[l + 32] + b1[l + 32] + p[l + 32];
}

// ---------------- K4: iq/ik = (qk @ W^T + b)*scale, relayout to (bh,s,md)
// 64x64 weight staged in LDS via CDNA5 async-to-LDS DMA; blocks of 8 waves,
// wave = one m-tile, full N=64. 6400 m-tiles -> 800 blocks exactly.
__global__ void k_innerqk(const float* __restrict__ qk, const float* __restrict__ W,
                          const float* __restrict__ bias, float scale,
                          float* __restrict__ dst) {
  __shared__ __align__(16) float shW[MD * MD];          // 16 KB
  const int tid = threadIdx.x;
  {
    // low 32 bits of the flat shared-pointer == LDS byte offset (ISA 10.2)
    unsigned lds0 = (unsigned)(unsigned long long)(&shW[0]);
#pragma unroll
    for (int rnd = 0; rnd < 4; ++rnd) {                 // 4 x 256 lanes x 16B = 16 KB
      unsigned off = (unsigned)(rnd * 4096 + tid * 16);
      asm volatile("global_load_async_to_lds_b128 %0, %1, %2"
                   :: "v"(lds0 + off), "v"(off), "s"(W) : "memory");
    }
    asm volatile("s_wait_asynccnt 0x0" ::: "memory");
    __syncthreads();
  }
  int mt = blockIdx.x * WPB + (tid >> 5);
  const float* A = qk + mt * 16 * MD;
  v8f acc[4] = {};
  for (int k0 = 0; k0 < MD; k0 += 32) {
    v16bf a = load_frag(A, MD, 1, k0, MD);
#pragma unroll
    for (int j = 0; j < 4; ++j) {
      v16bf b = load_frag(&shW[j * 16 * MD], MD, 1, k0, MD);   // ds_load path
      acc[j] = wmma_acc(a, b, acc[j]);
    }
  }
  int m0 = mt * 16, s = m0 >> 8, bh0 = m0 & 255;        // row = s*BH + bh
#pragma unroll
  for (int j = 0; j < 4; ++j)
    store_tile_bs(acc[j], dst + (bh0 * L + s) * MD + j * 16, L * MD, bias, j * 16,
                  scale);
}

// ---------------- K5a: ia_raw[bh] = iq[bh] @ ik[bh]^T   (400x400x64)
__global__ void k_iqikt(const float* __restrict__ iq, const float* __restrict__ ik,
                        float* __restrict__ ia) {
  int w = wave_id();
  if (w >= BH * 25 * 7) return;
  int bh = w / 175, rem = w % 175, mt = rem / 7, ng = rem % 7;
  const float* A = iq + bh * (L * MD) + mt * 16 * MD;
  v8f acc[4] = {};
  for (int k0 = 0; k0 < MD; k0 += 32) {
    v16bf a = load_frag(A, MD, 1, k0, MD);
#pragma unroll
    for (int j = 0; j < 4; ++j) {
      if (ng * 4 + j < 25) {
        v16bf b = load_frag(ik + bh * (L * MD) + (ng * 64 + j * 16) * MD, MD, 1, k0, MD);
        acc[j] = wmma_acc(a, b, acc[j]);
      }
    }
  }
#pragma unroll
  for (int j = 0; j < 4; ++j)
    if (ng * 4 + j < 25)
      store_tile(acc[j], ia + bh * LL + mt * 16 * L + (ng * 64 + j * 16), L);
}

// ---------------- softmax over a contiguous row of 400 (wave per row, in place)
__global__ void k_softmax400(float* __restrict__ buf, int nrows) {
  int w = wave_id();
  if (w >= nrows) return;
  float* row = buf + (long)w * L;
  int l = threadIdx.x & 31;
  float mx = -3.0e38f;
  for (int s = l; s < L; s += 32) mx = fmaxf(mx, row[s]);
  mx = wmaxr(mx);
  float sum = 0.f;
  for (int s = l; s < L; s += 32) { float e = __expf(row[s] - mx); row[s] = e; sum += e; }
  sum = wsum(sum);
  float inv = 1.0f / sum;
  for (int s = l; s < L; s += 32) row[s] *= inv;
}

// ---------------- K6: io[bh] = ia[bh] @ iv[bh], LN2 fused into B-fragment load
// iv[bh,k,f] = (attn[bh,f,k]-mean2[bh,k])*rstd2[bh,k]*g2[f]+b2[f]
__global__ void k_innerav(const float* __restrict__ ia, const float* __restrict__ attn,
                          const float* __restrict__ mean2, const float* __restrict__ rstd2,
                          const float* __restrict__ g2, const float* __restrict__ b2,
                          float* __restrict__ io) {
  int w = wave_id();
  if (w >= BH * 25 * 7) return;
  int bh = w / 175, rem = w % 175, mt = rem / 7, ng = rem % 7;
  const float* Arow = ia + bh * LL + mt * 16 * L;
  const float* m2 = mean2 + bh * L;
  const float* r2 = rstd2 + bh * L;
  const int l = threadIdx.x & 31, r = l & 15, hi = l >> 4;
  float gf[4], bf[4];
  const float* acol[4];
#pragma unroll
  for (int j = 0; j < 4; ++j) {
    gf[j] = 0.f; bf[j] = 0.f; acol[j] = attn;
    if (ng * 4 + j < 25) {
      int f = (ng * 4 + j) * 16 + r;
      gf[j] = g2[f]; bf[j] = b2[f];
      acol[j] = attn + bh * LL + f * L;                 // attn[bh, f, k]
    }
  }
  v8f acc[4] = {};
  for (int k0 = 0; k0 < L; k0 += 32) {
    v16bf a = load_frag(Arow, L, 1, k0, L);
#pragma unroll
    for (int j = 0; j < 4; ++j) {
      if (ng * 4 + j < 25) {
        v16bf bb;
#pragma unroll
        for (int e = 0; e < 16; ++e) {
          int k = k0 + ((e < 8) ? e : (e + 8)) + hi * 8;
          float x = 0.0f;
          if (k < L) x = (acol[j][k] - m2[k]) * r2[k] * gf[j] + bf[j];
          bb[e] = (__bf16)x;
        }
        acc[j] = wmma_acc(a, bb, acc[j]);
      }
    }
  }
#pragma unroll
  for (int j = 0; j < 4; ++j)
    if (ng * 4 + j < 25)
      store_tile(acc[j], io + bh * LL + mt * 16 * L + (ng * 64 + j * 16), L);
}

// ---------------- K7: attn[bh,t,s] += io[bh,s,t] + (io[bh,s,:] @ Wi[t,:]) + bi[t]
__global__ void k_corrout(const float* __restrict__ io, const float* __restrict__ Wi,
                          const float* __restrict__ bi, float* __restrict__ attn) {
  int w = wave_id();
  if (w >= BH * 25 * 7) return;
  int bh = w / 175, rem = w % 175, mt = rem / 7, ng = rem % 7;   // mt: s, ng: t-group
  const float* A = io + bh * LL + mt * 16 * L;
  v8f acc[4] = {};
  for (int k0 = 0; k0 < FS; k0 += 32) {
    v16bf a = load_frag(A, L, 1, k0, FS);
#pragma unroll
    for (int j = 0; j < 4; ++j) {
      if (ng * 4 + j < 25) {
        v16bf b = load_frag(Wi + (ng * 64 + j * 16) * FS, FS, 1, k0, FS);
        acc[j] = wmma_acc(a, b, acc[j]);
      }
    }
  }
  const int l = threadIdx.x & 31, n = l & 15, mb = (l >> 4) * 8;
#pragma unroll
  for (int j = 0; j < 4; ++j) {
    if (ng * 4 + j < 25) {
      int t = (ng * 4 + j) * 16 + n;
      float bias = bi[t];
#pragma unroll
      for (int i = 0; i < 8; ++i) {
        int s = mt * 16 + mb + i;
        float val = acc[j][i] + io[bh * LL + s * L + t] + bias;
        float* p = attn + bh * LL + t * L + s;          // unique transposed RMW
        *p = *p + val;
      }
    }
  }
}

// ---------------- K9: ctx2[l,b,h*32+d] = softmax(attn)[bh] @ v[bh]   (400x32x400)
__global__ void k_wv(const float* __restrict__ attn, const float* __restrict__ v,
                     float* __restrict__ ctx2) {
  int w = wave_id();
  if (w >= BH * 25) return;
  int bh = w / 25, mt = w % 25;
  const float* A  = attn + bh * LL + mt * 16 * L;
  const float* Bt = v + bh * (L * HD);                  // B^T[d][k] = v[bh,k,d]
  v8f acc[2] = {};
  gemm_nt<2>(A, L, 1, Bt, 1, HD, L, acc);
  int b = bh >> 3, h = bh & 7;
#pragma unroll
  for (int j = 0; j < 2; ++j)
    store_tile(acc[j], ctx2 + ((mt * 16) * B + b) * E + h * HD + j * 16, B * E);
}

// ---------------- K10: out = ctx2 @ out_w^T + out_b   ((L*B) x E x E)
__global__ void k_outproj(const float* __restrict__ ctx2, const float* __restrict__ W,
                          const float* __restrict__ bias, float* __restrict__ out) {
  int w = wave_id();
  if (w >= 800 * 4) return;
  int mt = w / 4, ng = w % 4;
  const float* A  = ctx2 + mt * 16 * E;
  const float* Bt = W + (ng * 64) * E;
  v8f acc[4] = {};
  gemm_nt<4>(A, E, 1, Bt, E, 1, E, acc);
#pragma unroll
  for (int j = 0; j < 4; ++j)
    store_tile_bs(acc[j], out + mt * 16 * E + ng * 64 + j * 16, E, bias,
                  ng * 64 + j * 16, 1.0f);
}

// =====================================================================
extern "C" void kernel_launch(void* const* d_in, const int* in_sizes, int n_in,
                              void* d_out, int out_size, void* d_ws, size_t ws_size,
                              hipStream_t stream) {
  const float* query   = (const float*)d_in[0];
  const float* key     = (const float*)d_in[1];
  const float* value   = (const float*)d_in[2];
  const float* pos_emb = (const float*)d_in[3];
  const float* in_w    = (const float*)d_in[4];
  const float* in_b    = (const float*)d_in[5];
  const float* out_w   = (const float*)d_in[6];
  const float* out_b   = (const float*)d_in[7];
  const float* corr_w  = (const float*)d_in[8];
  const float* corr_b  = (const float*)d_in[9];
  const float* ln1_g   = (const float*)d_in[10];
  const float* ln1_b   = (const float*)d_in[11];
  const float* ln2_g   = (const float*)d_in[12];
  const float* ln2_b   = (const float*)d_in[13];
  const float* q_w     = (const float*)d_in[14];
  const float* q_b     = (const float*)d_in[15];
  const float* k_w     = (const float*)d_in[16];
  const float* k_b     = (const float*)d_in[17];
  const float* io_w    = (const float*)d_in[18];
  const float* io_b    = (const float*)d_in[19];

  float* ws = (float*)d_ws;
  const size_t QKV = (size_t)BH * L * HD;   // 3,276,800
  const size_t ATT = (size_t)BH * L * L;    // 40,960,000
  const size_t QKB = (size_t)L * BH * MD;   // 6,553,600
  float* qb    = ws;                 float* kb    = qb + QKV;
  float* vb    = kb + QKV;           float* attn  = vb + QKV;
  float* qk    = attn + ATT;         float* iq    = qk + QKB;
  float* ik    = iq + QKB;           float* ia    = ik + QKB;
  float* io    = ia + ATT;           float* mean2 = io + ATT;
  float* rstd2 = mean2 + (size_t)BH * L;
  float* ctx2  = rstd2 + (size_t)BH * L;

  auto blocks = [](long waves) { return dim3((unsigned)((waves + WPB - 1) / WPB)); };
  const dim3 tb(WPB * 32);
  const float qscale  = 0.1767766952966369f;   // HD^-0.5
  const float iqscale = 0.125f;                // MD^-0.5

  // 1) fused in-projection (q scaled)
  k_inproj<<<blocks(3200), tb, 0, stream>>>(query, in_w,             in_b,         qb, qscale);
  k_inproj<<<blocks(3200), tb, 0, stream>>>(key,   in_w + E * E,     in_b + E,     kb, 1.0f);
  k_inproj<<<blocks(3200), tb, 0, stream>>>(value, in_w + 2 * E * E, in_b + 2 * E, vb, 1.0f);
  // 2) correlation map
  k_qkt<<<blocks(44800), tb, 0, stream>>>(qb, kb, attn);
  // 3) corr projection + LN stats + LN1 + pos
  k_corrproj<<<blocks(6400), tb, 0, stream>>>(attn, corr_w, corr_b, qk);
  k_ln2stats<<<blocks(102400), tb, 0, stream>>>(attn, mean2, rstd2);
  k_ln1pos<<<blocks(102400), tb, 0, stream>>>(qk, ln1_g, ln1_b, pos_emb);
  // 4) inner q/k projections (async-to-LDS staged weights)
  k_innerqk<<<dim3(800), tb, 0, stream>>>(qk, q_w, q_b, iqscale, iq);
  k_innerqk<<<dim3(800), tb, 0, stream>>>(qk, k_w, k_b, 1.0f,    ik);
  // 5) inner attention weights
  k_iqikt<<<blocks(44800), tb, 0, stream>>>(iq, ik, ia);
  k_softmax400<<<blocks(102400), tb, 0, stream>>>(ia, BH * L);
  // 6) io = ia @ LN2(cm)  (LN fused into B-fragment load)
  k_innerav<<<blocks(44800), tb, 0, stream>>>(ia, attn, mean2, rstd2, ln2_g, ln2_b, io);
  // 7) inner out-proj residual, scatter-added transposed into attn logits
  k_corrout<<<blocks(44800), tb, 0, stream>>>(io, io_w, io_b, attn);
  // 8) outer softmax + context + out-proj
  k_softmax400<<<blocks(102400), tb, 0, stream>>>(attn, BH * L);
  k_wv<<<blocks(6400), tb, 0, stream>>>(attn, vb, ctx2);
  k_outproj<<<blocks(3200), tb, 0, stream>>>(ctx2, out_w, out_b, (float*)d_out);
}
